// LSTM_78872779423961
// MI455X (gfx1250) — matrix operand order repaired and checked by reading, not compiled, not run on hardware
//
#include <hip/hip_runtime.h>

// ---------------------------------------------------------------------------
// 3-layer LSTM, T=1000, B=256, H=1024, persistent WMMA kernel for gfx1250.
// ---------------------------------------------------------------------------

#define T_STEPS 1000
#define BATCH   256
#define HID     1024
#define FOURH   4096
#define NBLK    64          // persistent grid: 64 blocks x 256 threads (8 waves)

typedef __attribute__((ext_vector_type(16))) __bf16 v16bf;
typedef __attribute__((ext_vector_type(8)))  float  v8f;

union FragU { struct { uint4 lo, hi; } q; v16bf v; };

// fp32 -> bf16 (round to nearest even), stored as raw u16
__device__ __forceinline__ unsigned short f2bf(float f) {
    unsigned u = __float_as_uint(f);
    unsigned r = ((u >> 16) & 1u) + 0x7FFFu;
    return (unsigned short)((u + r) >> 16);
}
__device__ __forceinline__ float bf2f(unsigned short u) {
    return __uint_as_float(((unsigned)u) << 16);
}

// Branch-free activations: v_exp_f32 + v_rcp_f32, no EXEC divergence.
__device__ __forceinline__ float fast_sigmoid(float v) {
    return __builtin_amdgcn_rcpf(1.0f + __expf(-v));
}
__device__ __forceinline__ float fast_tanh(float v) {
    // tanh(x) = 1 - 2/(1+exp(2x)); exp->inf / exp->0 limits give +/-1 exactly.
    return 1.0f - 2.0f * __builtin_amdgcn_rcpf(1.0f + __expf(2.0f * v));
}

// A fragment (16x32 bf16): lane holds row (lane&15); K chunks at base=(lane>>4)*8
// covering [base, base+8) and [base+16, base+24).  Row-major src, ldk = HID.
__device__ __forceinline__ v16bf load_a(const unsigned short* base, int row0,
                                        int k0, int lane) {
    const unsigned short* p = base + (size_t)(row0 + (lane & 15)) * HID
                              + k0 + ((lane >> 4) << 3);
    FragU u;
    u.q.lo = *reinterpret_cast<const uint4*>(p);
    u.q.hi = *reinterpret_cast<const uint4*>(p + 16);
    return u.v;
}

// B fragment (32x16 bf16): lane holds column (lane&15) == weight row; K chunks at
// base=(lane>>4)*8 covering [base, base+8) and [base+16, base+24).  Weight is row-major [4H, HID].
__device__ __forceinline__ v16bf load_b(const unsigned short* base, int row0,
                                        int k0, int lane) {
    const unsigned short* p = base + (size_t)(row0 + (lane & 15)) * HID
                              + k0 + ((lane >> 4) << 3);
    FragU u;
    u.q.lo = *reinterpret_cast<const uint4*>(p);
    u.q.hi = *reinterpret_cast<const uint4*>(p + 16);
    return u.v;
}

#define WMMA_BF16(A, B, C) \
    __builtin_amdgcn_wmma_f32_16x16x32_bf16(false, (A), false, (B), (short)0, (C), false, false)

// ---------------------------------------------------------------------------
// Grid-wide barrier (atomic counter + generation, agent scope)
// ---------------------------------------------------------------------------
__device__ __forceinline__ void gsync(unsigned* cnt, unsigned* gen, unsigned nb) {
    __syncthreads();
    __threadfence();
    if (threadIdx.x == 0) {
        unsigned g = __hip_atomic_load(gen, __ATOMIC_RELAXED, __HIP_MEMORY_SCOPE_AGENT);
        unsigned a = __hip_atomic_fetch_add(cnt, 1u, __ATOMIC_ACQ_REL, __HIP_MEMORY_SCOPE_AGENT);
        if (a == nb - 1u) {
            __hip_atomic_store(cnt, 0u, __ATOMIC_RELAXED, __HIP_MEMORY_SCOPE_AGENT);
            __hip_atomic_store(gen, g + 1u, __ATOMIC_RELEASE, __HIP_MEMORY_SCOPE_AGENT);
        } else {
            while (__hip_atomic_load(gen, __ATOMIC_ACQUIRE, __HIP_MEMORY_SCOPE_AGENT) == g) {
                __builtin_amdgcn_s_sleep(8);
            }
        }
    }
    __syncthreads();
    __threadfence();
}

// ---------------------------------------------------------------------------
// Init: weights fp32->bf16, fused biases, seed h/c buffers, reset barrier.
// ---------------------------------------------------------------------------
__global__ void lstm_init(const float* __restrict__ Whh1f,
                          const float* __restrict__ Wih2f,
                          const float* __restrict__ Whh2f,
                          const float* __restrict__ hx1,
                          const float* __restrict__ hx2,
                          const float* __restrict__ bih1, const float* __restrict__ bhh1,
                          const float* __restrict__ Wih1,
                          const float* __restrict__ bih2, const float* __restrict__ bhh2,
                          const float* __restrict__ hx3,  const float* __restrict__ cx3,
                          unsigned short* __restrict__ Whh1b,
                          unsigned short* __restrict__ Wih2b,
                          unsigned short* __restrict__ Whh2b,
                          unsigned short* __restrict__ h1buf,
                          unsigned short* __restrict__ h2buf,
                          float* __restrict__ bias1, float* __restrict__ wcol1,
                          float* __restrict__ bias2,
                          float* __restrict__ h3buf, float* __restrict__ c3buf,
                          unsigned* __restrict__ syncp) {
    int i = blockIdx.x * 256 + threadIdx.x;
    if (i < FOURH * HID) {
        Whh1b[i] = f2bf(Whh1f[i]);
        Wih2b[i] = f2bf(Wih2f[i]);
        Whh2b[i] = f2bf(Whh2f[i]);
    }
    if (i < BATCH * HID) {
        h1buf[i] = f2bf(hx1[i]);   // ping buffer 0 holds h at t=0
        h2buf[i] = f2bf(hx2[i]);
    }
    if (i < FOURH) {
        bias1[i] = bih1[i] + bhh1[i];
        wcol1[i] = Wih1[i];                     // W_ih1 is [4H,1]
        bias2[i] = bih2[i] + bhh2[i];
    }
    if (i < BATCH) { h3buf[i] = hx3[i]; c3buf[i] = cx3[i]; }
    if (i < 2) syncp[i] = 0u;
}

// ---------------------------------------------------------------------------
// Persistent kernel: whole T-loop, 3 grid syncs per step.
// Block b = hidden tile (16 cols), wave w = batch tile (32 rows).
// Each wave: 32x16x(4 gates) output tile, c-state kept in registers.
// ---------------------------------------------------------------------------
__global__ __launch_bounds__(256, 1)
void lstm_persistent(const float* __restrict__ x,
                     const float* __restrict__ cx1, const float* __restrict__ cx2,
                     const unsigned short* __restrict__ Whh1,
                     const unsigned short* __restrict__ Wih2,
                     const unsigned short* __restrict__ Whh2,
                     const float* __restrict__ bias1, const float* __restrict__ wcol1,
                     const float* __restrict__ bias2,
                     const float* __restrict__ Wih3, const float* __restrict__ Whh3,
                     const float* __restrict__ bih3, const float* __restrict__ bhh3,
                     unsigned short* __restrict__ h1buf,   // 2 x [256,1024] bf16
                     unsigned short* __restrict__ h2buf,   // 2 x [256,1024] bf16
                     float* __restrict__ h3buf, float* __restrict__ c3buf,
                     float* __restrict__ out,              // [1000,256]
                     unsigned* __restrict__ syncp) {
    __shared__ float red[256];
    __shared__ float gatsh[16][4];

    const int tid   = threadIdx.x;
    const int lane  = tid & 31;
    const int wave  = tid >> 5;
    const int jtile = blockIdx.x;       // 0..63 hidden tile
    const int mtile = wave;             // 0..7 batch tile
    const int j0 = jtile * 16;
    const int m0 = mtile * 32;
    const int hi16 = lane >> 4;
    const int jn = lane & 15;
    const int j = j0 + jn;

    unsigned* cnt = syncp;
    unsigned* gen = syncp + 1;

    // Hoisted per-lane gate constants
    float b1c[4], w1c[4], b2c[4];
#pragma unroll
    for (int g = 0; g < 4; ++g) {
        b1c[g] = bias1[g * HID + j];
        w1c[g] = wcol1[g * HID + j];
        b2c[g] = bias2[g * HID + j];
    }

    // Cell states in registers: element (ms,r) -> batch m0+ms*16+hi16*8+r, col j
    float c1s[16], c2s[16];
#pragma unroll
    for (int ms = 0; ms < 2; ++ms)
#pragma unroll
        for (int r = 0; r < 8; ++r) {
            int m = m0 + ms * 16 + hi16 * 8 + r;
            c1s[ms * 8 + r] = cx1[(size_t)m * HID + j];
            c2s[ms * 8 + r] = cx2[(size_t)m * HID + j];
        }

    const v8f vzero = {0.f, 0.f, 0.f, 0.f, 0.f, 0.f, 0.f, 0.f};

#pragma unroll 1
    for (int t = 0; t < T_STEPS; ++t) {
        const unsigned short* h1p = h1buf + (size_t)(t & 1) * (BATCH * HID);
        unsigned short*       h1n = h1buf + (size_t)((t + 1) & 1) * (BATCH * HID);
        const unsigned short* h2p = h2buf + (size_t)(t & 1) * (BATCH * HID);
        unsigned short*       h2n = h2buf + (size_t)((t + 1) & 1) * (BATCH * HID);

        // ------------------ Layer 1: gates = h1 @ Whh1^T (+ x*wcol + b) -----
        v8f acc[8];
#pragma unroll
        for (int q = 0; q < 8; ++q) acc[q] = vzero;

#pragma unroll 2
        for (int k0 = 0; k0 < HID; k0 += 32) {
            v16bf a0 = load_a(h1p, m0,      k0, lane);
            v16bf a1 = load_a(h1p, m0 + 16, k0, lane);
#pragma unroll
            for (int g = 0; g < 4; ++g) {
                v16bf bb = load_b(Whh1, (g << 10) + j0, k0, lane);
                acc[2 * g]     = WMMA_BF16(a0, bb, acc[2 * g]);
                acc[2 * g + 1] = WMMA_BF16(a1, bb, acc[2 * g + 1]);
            }
        }
#pragma unroll
        for (int ms = 0; ms < 2; ++ms)
#pragma unroll
            for (int r = 0; r < 8; ++r) {
                int m = m0 + ms * 16 + hi16 * 8 + r;
                float xv = x[t * BATCH + m];
                float iv = fast_sigmoid(acc[0 + ms][r] + b1c[0] + xv * w1c[0]);
                float fv = fast_sigmoid(acc[2 + ms][r] + b1c[1] + xv * w1c[1]);
                float gv = fast_tanh   (acc[4 + ms][r] + b1c[2] + xv * w1c[2]);
                float ov = fast_sigmoid(acc[6 + ms][r] + b1c[3] + xv * w1c[3]);
                float cc = fv * c1s[ms * 8 + r] + iv * gv;
                c1s[ms * 8 + r] = cc;
                h1n[(size_t)m * HID + j] = f2bf(ov * fast_tanh(cc));
            }
        gsync(cnt, gen, NBLK);

        // ------------- Layer 2: gates = h1_t @ Wih2^T + h2 @ Whh2^T + b -----
#pragma unroll
        for (int q = 0; q < 8; ++q) acc[q] = vzero;

#pragma unroll 2
        for (int k0 = 0; k0 < HID; k0 += 32) {
            v16bf a0 = load_a(h1n, m0,      k0, lane);
            v16bf a1 = load_a(h1n, m0 + 16, k0, lane);
#pragma unroll
            for (int g = 0; g < 4; ++g) {
                v16bf bb = load_b(Wih2, (g << 10) + j0, k0, lane);
                acc[2 * g]     = WMMA_BF16(a0, bb, acc[2 * g]);
                acc[2 * g + 1] = WMMA_BF16(a1, bb, acc[2 * g + 1]);
            }
        }
#pragma unroll 2
        for (int k0 = 0; k0 < HID; k0 += 32) {
            v16bf a0 = load_a(h2p, m0,      k0, lane);
            v16bf a1 = load_a(h2p, m0 + 16, k0, lane);
#pragma unroll
            for (int g = 0; g < 4; ++g) {
                v16bf bb = load_b(Whh2, (g << 10) + j0, k0, lane);
                acc[2 * g]     = WMMA_BF16(a0, bb, acc[2 * g]);
                acc[2 * g + 1] = WMMA_BF16(a1, bb, acc[2 * g + 1]);
            }
        }
#pragma unroll
        for (int ms = 0; ms < 2; ++ms)
#pragma unroll
            for (int r = 0; r < 8; ++r) {
                int m = m0 + ms * 16 + hi16 * 8 + r;
                float iv = fast_sigmoid(acc[0 + ms][r] + b2c[0]);
                float fv = fast_sigmoid(acc[2 + ms][r] + b2c[1]);
                float gv = fast_tanh   (acc[4 + ms][r] + b2c[2]);
                float ov = fast_sigmoid(acc[6 + ms][r] + b2c[3]);
                float cc = fv * c2s[ms * 8 + r] + iv * gv;
                c2s[ms * 8 + r] = cc;
                h2n[(size_t)m * HID + j] = f2bf(ov * fast_tanh(cc));
            }
        gsync(cnt, gen, NBLK);

        // -------- Layer 3 (H=1024 -> 1): 16 blocks, LDS reduction ----------
        if (blockIdx.x < 16) {
            int pair = tid >> 2, ks = tid & 3;          // 64 (row,gate) pairs x 4 K-slices
            int row = pair >> 2, g3 = pair & 3;
            int bat = blockIdx.x * 16 + row;
            const uint4*  hv = reinterpret_cast<const uint4*>(h2n + (size_t)bat * HID + ks * 256);
            const float4* wv = reinterpret_cast<const float4*>(Wih3 + g3 * HID + ks * 256);
            float s = 0.f;
#pragma unroll 4
            for (int k = 0; k < 32; ++k) {              // 8 bf16 + 8 fp32 per iter
                uint4  hq = hv[k];
                float4 w0 = wv[2 * k];
                float4 w1 = wv[2 * k + 1];
                s += bf2f((unsigned short)(hq.x & 0xffffu)) * w0.x;
                s += bf2f((unsigned short)(hq.x >> 16))     * w0.y;
                s += bf2f((unsigned short)(hq.y & 0xffffu)) * w0.z;
                s += bf2f((unsigned short)(hq.y >> 16))     * w0.w;
                s += bf2f((unsigned short)(hq.z & 0xffffu)) * w1.x;
                s += bf2f((unsigned short)(hq.z >> 16))     * w1.y;
                s += bf2f((unsigned short)(hq.w & 0xffffu)) * w1.z;
                s += bf2f((unsigned short)(hq.w >> 16))     * w1.w;
            }
            red[tid] = s;
            __syncthreads();
            if (ks == 0) {
                float tot = red[tid] + red[tid + 1] + red[tid + 2] + red[tid + 3]
                          + bih3[g3] + bhh3[g3] + h3buf[bat] * Whh3[g3];
                gatsh[row][g3] = tot;
            }
            __syncthreads();
            if (tid < 16) {
                int bb = blockIdx.x * 16 + tid;
                float iv = fast_sigmoid(gatsh[tid][0]);
                float fv = fast_sigmoid(gatsh[tid][1]);
                float gv = fast_tanh   (gatsh[tid][2]);
                float ov = fast_sigmoid(gatsh[tid][3]);
                float cc = fv * c3buf[bb] + iv * gv;
                c3buf[bb] = cc;
                float hh = ov * fast_tanh(cc);
                h3buf[bb] = hh;
                out[t * BATCH + bb] = hh;
            }
        }
        gsync(cnt, gen, NBLK);
    }
}

// ---------------------------------------------------------------------------
extern "C" void kernel_launch(void* const* d_in, const int* in_sizes, int n_in,
                              void* d_out, int out_size, void* d_ws, size_t ws_size,
                              hipStream_t stream) {
    (void)in_sizes; (void)n_in; (void)out_size; (void)ws_size;

    const float* x     = (const float*)d_in[0];
    const float* hx1   = (const float*)d_in[1];
    const float* cx1   = (const float*)d_in[2];
    const float* hx2   = (const float*)d_in[3];
    const float* cx2   = (const float*)d_in[4];
    const float* hx3   = (const float*)d_in[5];
    const float* cx3   = (const float*)d_in[6];
    const float* Wih1  = (const float*)d_in[7];
    const float* Whh1f = (const float*)d_in[8];
    const float* bih1  = (const float*)d_in[9];
    const float* bhh1  = (const float*)d_in[10];
    const float* Wih2f = (const float*)d_in[11];
    const float* Whh2f = (const float*)d_in[12];
    const float* bih2  = (const float*)d_in[13];
    const float* bhh2  = (const float*)d_in[14];
    const float* Wih3  = (const float*)d_in[15];
    const float* Whh3  = (const float*)d_in[16];
    const float* bih3  = (const float*)d_in[17];
    const float* bhh3  = (const float*)d_in[18];

    char* ws = (char*)d_ws;
    size_t off = 0;
    unsigned* syncp       = (unsigned*)(ws + off);       off += 256;
    unsigned short* Whh1b = (unsigned short*)(ws + off); off += (size_t)FOURH * HID * 2;
    unsigned short* Wih2b = (unsigned short*)(ws + off); off += (size_t)FOURH * HID * 2;
    unsigned short* Whh2b = (unsigned short*)(ws + off); off += (size_t)FOURH * HID * 2;
    float* bias1          = (float*)(ws + off);          off += FOURH * 4;
    float* wcol1          = (float*)(ws + off);          off += FOURH * 4;
    float* bias2          = (float*)(ws + off);          off += FOURH * 4;
    unsigned short* h1buf = (unsigned short*)(ws + off); off += (size_t)2 * BATCH * HID * 2;
    unsigned short* h2buf = (unsigned short*)(ws + off); off += (size_t)2 * BATCH * HID * 2;
    float* h3buf          = (float*)(ws + off);          off += 1024;
    float* c3buf          = (float*)(ws + off);          off += 1024;

    float* out = (float*)d_out;

    const int initN = FOURH * HID;                  // 4,194,304
    lstm_init<<<dim3((initN + 255) / 256), dim3(256), 0, stream>>>(
        Whh1f, Wih2f, Whh2f, hx1, hx2, bih1, bhh1, Wih1, bih2, bhh2, hx3, cx3,
        Whh1b, Wih2b, Whh2b, h1buf, h2buf, bias1, wcol1, bias2, h3buf, c3buf, syncp);

    lstm_persistent<<<dim3(NBLK), dim3(256), 0, stream>>>(
        x, cx1, cx2, Whh1b, Wih2b, Whh2b, bias1, wcol1, bias2,
        Wih3, Whh3, bih3, bhh3, h1buf, h2buf, h3buf, c3buf, out, syncp);
}
